// Generator_31774168056152
// MI455X (gfx1250) — matrix operand order repaired and checked by reading, not compiled, not run on hardware
//
#include <hip/hip_runtime.h>
#include <hip/hip_bf16.h>

typedef __bf16 v16bf __attribute__((ext_vector_type(16)));
typedef float  v8f   __attribute__((ext_vector_type(8)));

#define Bsz 4096
#define Tsz 512
#define Hsz 256
#define BT  (4096 * 512)

#if __has_builtin(__builtin_amdgcn_tanh_f32)
__device__ __forceinline__ float fast_tanh(float v) { return __builtin_amdgcn_tanh_f32(v); }
#elif __has_builtin(__builtin_amdgcn_tanhf)
__device__ __forceinline__ float fast_tanh(float v) { return __builtin_amdgcn_tanhf(v); }
#else
__device__ __forceinline__ float fast_tanh(float v) { return tanhf(v); }
#endif
// sigmoid(x) = 0.5*tanh(x/2) + 0.5  (one TRANS op on CDNA5)
__device__ __forceinline__ float fast_sigmoid(float v) {
  return __builtin_fmaf(0.5f, fast_tanh(0.5f * v), 0.5f);
}

// ---------------------------------------------------------------------------
// Repack weights -> bf16 WMMA B fragments in d_ws (65 nTiles x 8 kTiles).
//   nTile 0..63 : w_hh.T   (N = gate-output col, K = hidden)
//   nTile 64    : head tile (col 0 = w_mean, col 1 = w_logvar, cols 2..15 = 0)
// B layout (32x16, bf16): lane = N (lane&15) + 16*h, element s = K in table order.
// ---------------------------------------------------------------------------
__global__ __launch_bounds__(256) void repack_weights_kernel(
    const float* __restrict__ whh, const float* __restrict__ wmean,
    const float* __restrict__ wlogvar, unsigned short* __restrict__ wrep_raw) {
  int tid = blockIdx.x * 256 + threadIdx.x;
  if (tid >= 65 * 8 * 32 * 16) return;
  int frag  = tid >> 9;       // /512
  int rem   = tid & 511;
  int laneK = rem >> 4;
  int n     = laneK & 15;
  int s     = rem & 15;
  int nTile = frag >> 3;
  int kt    = frag & 7;
  int k     = kt * 32 + (s >> 3) * 16 + (laneK >> 4) * 8 + (s & 7);
  float v;
  if (nTile < 64) {
    v = whh[(nTile * 16 + n) * Hsz + k];
  } else {
    v = (n == 0) ? wmean[k] : (n == 1) ? wlogvar[k] : 0.0f;
  }
  ((__bf16*)wrep_raw)[tid] = (__bf16)v;
}

// ---------------------------------------------------------------------------
// Persistent LSTM: one WG = 16 batch rows, 8 waves. Wave w owns jt = 2w, 2w+1;
// per jt it computes the four gate tiles (i,f,g,o) with
// V_WMMA_F32_16X16X32_BF16, does the pointwise cell update in-register, and
// scatters bf16 h into the LDS A-fragment buffer for the next timestep.
// Wave 0 additionally multiplies A (= h_{t-1}) by the head tile, so the
// mean/logvar dots come out of the matrix pipe; outputs lag one iteration.
// ---------------------------------------------------------------------------
__global__ __launch_bounds__(256) void lstm_wmma_kernel(
    const float* __restrict__ x, const float* __restrict__ eps,
    const float* __restrict__ w_ih, const float* __restrict__ b_ih,
    const float* __restrict__ b_hh, const float* __restrict__ b_mean,
    const float* __restrict__ b_logvar,
    const unsigned short* __restrict__ wrep_raw, float* __restrict__ out) {
  __shared__ v16bf hA[8][32];      // A fragments of h (bf16), [kTile][lane]
  __shared__ float headLds[16][2]; // [row][0]=mean dot, [1]=logvar dot
  __shared__ float xLds[16];       // x_t per row

  const int tid    = threadIdx.x;
  const int lane   = tid & 31;
  const int wv     = tid >> 5;
  const int base   = blockIdx.x * 16;            // batch rows of this WG
  const int rowadd = (lane < 16) ? 0 : 8;        // C-matrix row mapping

  // zero the h A-fragment buffer (h0 = 0)
  {
    unsigned* p = (unsigned*)hA;                 // 8*32*16 bf16 = 2048 dwords
    for (int i = tid; i < 2048; i += 256) p[i] = 0u;
  }

  float c[2][8];                                 // cell state (c0 = 0)
#pragma unroll
  for (int q = 0; q < 2; ++q)
#pragma unroll
    for (int r = 0; r < 8; ++r) c[q][r] = 0.0f;

  // Per-lane invariants: hidden column j per q, input weights / combined
  // biases for the 4 gates, and the 16-bit A-layout slot (ISA 7.12.2) for
  // writing h back.
  float wihq[2][4], biasq[2][4];
  int ktA[2], slA[2], hb16[2];
#pragma unroll
  for (int q = 0; q < 2; ++q) {
    int jt  = wv * 2 + q;
    int j   = jt * 16 + (lane & 15);
    int kin = j & 31;
    ktA[q]  = j >> 5;
    hb16[q] = 16 * ((kin >> 3) & 1);
    slA[q]  = ((kin >> 4) * 4 + ((kin & 7) >> 1)) * 2 + (kin & 1);
#pragma unroll
    for (int g = 0; g < 4; ++g) {
      int col     = g * 256 + j;
      wihq[q][g]  = w_ih[col];
      biasq[q][g] = b_ih[col] + b_hh[col];
    }
  }
  const float bm = b_mean[0], blv = b_logvar[0];
  __syncthreads();

  // T+1 iterations: iteration t does the gate GEMM/pointwise for step t and
  // the head GEMM + output store for step t-1 (A fragments hold h_{t-1}).
  for (int t = 0; t <= Tsz; ++t) {
    // Launder a zero offset: keeps the global address space of wrep_raw but
    // makes the fragment addresses loop-variant so LICM cannot hoist ~512
    // VGPRs worth of loads out of the loop (which would spill to scratch).
    unsigned woff = 0;
    asm volatile("" : "+s"(woff));
    const v16bf* bp = (const v16bf*)(wrep_raw + woff);

    if (t < Tsz && tid < 16) xLds[tid] = x[(size_t)(base + tid) * Tsz + t];

    // Load A fragments (h_{t-1}): 8 x v16bf per lane, shared by all waves.
    v16bf a[8];
#pragma unroll
    for (int k = 0; k < 8; ++k) a[k] = hA[k][lane];
    __syncthreads();  // all A reads + xLds done before anyone rewrites hA

    // ---- head GEMM (wave 0 only): [mean|logvar] dots of h_{t-1} ----
    if (wv == 0 && t > 0) {
      v8f aH = {0,0,0,0,0,0,0,0};
#pragma unroll
      for (int k = 0; k < 8; ++k) {
        v16bf bH = bp[((64 * 8 + k) * 32) + lane];
        aH = __builtin_amdgcn_wmma_f32_16x16x32_bf16(false, a[k], false, bH,
                                                     (short)0, aH, false, false);
      }
      int cidx = lane & 15;   // C layout: col = lane&15, row = r + rowadd
      if (cidx < 2) {
#pragma unroll
        for (int r = 0; r < 8; ++r) headLds[r + rowadd][cidx] = aH[r];
      }
    }

    // ---- gate GEMM + pointwise cell update for step t ----
    if (t < Tsz) {
#pragma unroll
      for (int q = 0; q < 2; ++q) {
        const int jt = wv * 2 + q;
        v8f aI = {0,0,0,0,0,0,0,0};
        v8f aF = {0,0,0,0,0,0,0,0};
        v8f aG = {0,0,0,0,0,0,0,0};
        v8f aO = {0,0,0,0,0,0,0,0};
#pragma unroll
        for (int k = 0; k < 8; ++k) {
          v16bf bI = bp[(((0 * 16 + jt) * 8 + k) * 32) + lane];
          v16bf bF = bp[(((1 * 16 + jt) * 8 + k) * 32) + lane];
          v16bf bG = bp[(((2 * 16 + jt) * 8 + k) * 32) + lane];
          v16bf bO = bp[(((3 * 16 + jt) * 8 + k) * 32) + lane];
          aI = __builtin_amdgcn_wmma_f32_16x16x32_bf16(false, a[k], false, bI,
                                                       (short)0, aI, false, false);
          aF = __builtin_amdgcn_wmma_f32_16x16x32_bf16(false, a[k], false, bF,
                                                       (short)0, aF, false, false);
          aG = __builtin_amdgcn_wmma_f32_16x16x32_bf16(false, a[k], false, bG,
                                                       (short)0, aG, false, false);
          aO = __builtin_amdgcn_wmma_f32_16x16x32_bf16(false, a[k], false, bO,
                                                       (short)0, aO, false, false);
        }

        // All 4 gate values for (row, j) live in this lane at acc index r.
        __bf16* hrow = (__bf16*)&hA[ktA[q]][0];
#pragma unroll
        for (int r = 0; r < 8; ++r) {
          int   row = r + rowadd;
          float xv  = xLds[row];
          float iv  = fast_sigmoid(aI[r] + xv * wihq[q][0] + biasq[q][0]);
          float fv  = fast_sigmoid(aF[r] + xv * wihq[q][1] + biasq[q][1]);
          float gv  = fast_tanh   (aG[r] + xv * wihq[q][2] + biasq[q][2]);
          float ov  = fast_sigmoid(aO[r] + xv * wihq[q][3] + biasq[q][3]);
          float cv  = fv * c[q][r] + iv * gv;
          c[q][r]   = cv;
          float hv  = ov * fast_tanh(cv);
          // scatter new h (bf16) into next step's A fragment
          hrow[(row + hb16[q]) * 16 + slA[q]] = (__bf16)hv;
        }
      }
    }
    __syncthreads();  // headLds and hA visible

    if (t > 0 && tid < 16) {  // heads + reparam sample for step t-1
      float  mean = headLds[tid][0] + bm;
      float  lv   = fminf(fmaxf(headLds[tid][1] + blv, -10.0f), 2.0f);
      size_t idx  = (size_t)(base + tid) * Tsz + (t - 1);
      float  e    = eps[idx];
      out[idx]                   = mean + __expf(0.5f * lv) * e;
      out[(size_t)BT + idx]      = mean;
      out[2 * (size_t)BT + idx]  = lv;
    }
  }
}

extern "C" void kernel_launch(void* const* d_in, const int* in_sizes, int n_in,
                              void* d_out, int out_size, void* d_ws, size_t ws_size,
                              hipStream_t stream) {
  const float* x        = (const float*)d_in[0];
  const float* eps      = (const float*)d_in[1];
  const float* w_ih     = (const float*)d_in[2];
  const float* w_hh     = (const float*)d_in[3];
  const float* b_ih     = (const float*)d_in[4];
  const float* b_hh     = (const float*)d_in[5];
  const float* w_mean   = (const float*)d_in[6];
  const float* b_mean   = (const float*)d_in[7];
  const float* w_logvar = (const float*)d_in[8];
  const float* b_logvar = (const float*)d_in[9];
  float*          out  = (float*)d_out;
  unsigned short* wrep = (unsigned short*)d_ws;   // 520 KiB bf16 weight tiles

  repack_weights_kernel<<<(65 * 8 * 32 * 16 + 255) / 256, 256, 0, stream>>>(
      w_hh, w_mean, w_logvar, wrep);
  lstm_wmma_kernel<<<Bsz / 16, 256, 0, stream>>>(
      x, eps, w_ih, b_ih, b_hh, b_mean, b_logvar, wrep, out);
}
